// VMFLoss_19696720020068
// MI455X (gfx1250) — matrix operand order repaired and checked by reading, not compiled, not run on hardware
//
#include <hip/hip_runtime.h>
#include <hip/hip_bf16.h>
#include <math.h>

#define DIM 512
#define NROWS 8192
#define BT 4                       // 4x4 sub-tiles of 16x16 => 64x64 per wave
#define NST (NROWS / 64)           // 128 super-tiles per matrix dimension
#define WAVES_PER_BLOCK 8

typedef __attribute__((ext_vector_type(16))) __bf16 v16bf;
typedef __attribute__((ext_vector_type(8)))  float  v8f;

// ---------------------------------------------------------------------------
// Kernel 0: normalize mu into ws, zero the 4 accumulators (replay-safe init).
// acc[0]=sum(norm) acc[1]=sum(norm^2) acc[2]=sum(dot) acc[3]=repulsion sum
// ---------------------------------------------------------------------------
__global__ __launch_bounds__(512)
void vmf_init_kernel(const float* __restrict__ mu, float* __restrict__ mu_n,
                     float* __restrict__ acc) {
    __shared__ float sdata[512];
    __shared__ float s_inv;
    const int t = threadIdx.x;
    float v = mu[t];
    sdata[t] = v * v;
    __syncthreads();
    for (int s = 256; s > 0; s >>= 1) {
        if (t < s) sdata[t] += sdata[t + s];
        __syncthreads();
    }
    if (t == 0) {
        float nrm = sqrtf(sdata[0]);
        s_inv = 1.0f / fmaxf(nrm, 1e-12f);
        acc[0] = 0.f; acc[1] = 0.f; acc[2] = 0.f; acc[3] = 0.f;
    }
    __syncthreads();
    mu_n[t] = v * s_inv;
}

// ---------------------------------------------------------------------------
// Kernel 1: per row -> norm, bf16 unit row, nsq (fp32), dot with mu_n.
// One 256-thread block per row (2 elements / thread).
// ---------------------------------------------------------------------------
__global__ __launch_bounds__(256)
void vmf_prep_kernel(const float* __restrict__ x, const float* __restrict__ mu_n,
                     __hip_bfloat16* __restrict__ xu, float* __restrict__ nsq,
                     float* __restrict__ acc) {
    __shared__ float sdata[256];
    __shared__ float s_inv;
    const int row = blockIdx.x;
    const int t   = threadIdx.x;
    const size_t base = (size_t)row * DIM;

    float x0 = x[base + t];
    float x1 = x[base + t + 256];
    sdata[t] = x0 * x0 + x1 * x1;
    __syncthreads();
    for (int s = 128; s > 0; s >>= 1) {
        if (t < s) sdata[t] += sdata[t + s];
        __syncthreads();
    }
    if (t == 0) {
        float sumsq = sdata[0];
        float nrm   = sqrtf(sumsq);
        float inv   = 1.0f / fmaxf(nrm, 1e-12f);
        s_inv = inv;
        nsq[row] = sumsq * inv * inv;           // ||x_unit||^2 (fp32, ~1)
        atomicAdd(&acc[0], nrm);
        atomicAdd(&acc[1], sumsq);
    }
    __syncthreads();                            // s_inv visible; sdata reusable
    float inv = s_inv;
    float u0 = x0 * inv, u1 = x1 * inv;
    xu[base + t]       = __float2bfloat16(u0);
    xu[base + t + 256] = __float2bfloat16(u1);
    sdata[t] = u0 * mu_n[t] + u1 * mu_n[t + 256];
    __syncthreads();
    for (int s = 128; s > 0; s >>= 1) {
        if (t < s) sdata[t] += sdata[t + s];
        __syncthreads();
    }
    if (t == 0) atomicAdd(&acc[2], sdata[0]);
}

// ---------------------------------------------------------------------------
// Kernel 2: Gram + repulsion, register-blocked + symmetric.
// One wave32 per 64x64 super-tile (upper triangle only): per k-step,
// 8 fragment loads feed 16 v_wmma_f32_16x16x32_bf16 (0.5 KB / WMMA).
// Off-diagonal super-tiles contribute twice (symmetry of 1/dist^2).
// ---------------------------------------------------------------------------
__global__ __launch_bounds__(WAVES_PER_BLOCK * 32)
void vmf_gram_kernel(const __hip_bfloat16* __restrict__ xu,
                     const float* __restrict__ nsq,
                     float* __restrict__ acc) {
    const int lane = threadIdx.x & 31;
    const int wave = threadIdx.x >> 5;
    const int w    = blockIdx.x * WAVES_PER_BLOCK + wave;
    const int tM   = w / NST;
    const int tN   = w % NST;
    if (tN < tM) return;                        // wave-uniform: EXEC stays all-1s
    const float weight = (tN == tM) ? 1.0f : 2.0f;

    const int mn = lane & 15;                   // A row / B col within sub-tile
    const int hv = lane >> 4;                   // half-wave selector

    const unsigned short* U = reinterpret_cast<const unsigned short*>(xu);

    v8f c[BT][BT];
    const v8f zero = {};
    #pragma unroll
    for (int r = 0; r < BT; ++r)
        #pragma unroll
        for (int s = 0; s < BT; ++s) c[r][s] = zero;

    for (int kk = 0; kk < DIM; kk += 32) {
        union Frag { v16bf v; uint4 u[2]; } a[BT], b[BT];
        #pragma unroll
        for (int r = 0; r < BT; ++r) {
            // A 16x32 bf16: lane holds row, K = hv*8+[0..7] and 16+hv*8+[0..7]
            const unsigned short* ar =
                U + (size_t)(tM * 64 + r * 16 + mn) * DIM + kk;
            a[r].u[0] = *reinterpret_cast<const uint4*>(ar + hv * 8);
            a[r].u[1] = *reinterpret_cast<const uint4*>(ar + 16 + hv * 8);
            // B 32x16 bf16: lane holds column (a row of U), K = hv*8+[0..7] and 16+hv*8+[0..7]
            const unsigned short* br =
                U + (size_t)(tN * 64 + r * 16 + mn) * DIM + kk;
            b[r].u[0] = *reinterpret_cast<const uint4*>(br + hv * 8);
            b[r].u[1] = *reinterpret_cast<const uint4*>(br + 16 + hv * 8);
        }
        #pragma unroll
        for (int r = 0; r < BT; ++r)
            #pragma unroll
            for (int s = 0; s < BT; ++s)
                c[r][s] = __builtin_amdgcn_wmma_f32_16x16x32_bf16(
                    false, a[r].v, false, b[s].v, (short)0, c[r][s], false, false);
    }

    // C/D layout: VGPR d -> M = d + 8*hv, N = lane&15
    float lsum = 0.0f;
    #pragma unroll
    for (int s = 0; s < BT; ++s) {
        const int gj = tN * 64 + s * 16 + mn;
        const float nsq_j = nsq[gj];
        #pragma unroll
        for (int r = 0; r < BT; ++r) {
            #pragma unroll
            for (int d = 0; d < 8; ++d) {
                int gi = tM * 64 + r * 16 + d + hv * 8;
                float dist = nsq[gi] + nsq_j - 2.0f * c[r][s][d] + 1e-6f;
                float invd = 1.0f / (dist + 1e-6f);
                lsum += (gi == gj) ? 0.0f : invd;   // exclude diagonal
            }
        }
    }
    lsum *= weight;
    // wave32 reduction
    #pragma unroll
    for (int off = 16; off > 0; off >>= 1)
        lsum += __shfl_xor(lsum, off, 32);
    if (lane == 0) atomicAdd(&acc[3], lsum);
}

// ---------------------------------------------------------------------------
// Kernel 3: scalar epilogue — fp32 log-Bessel series (matches reference's
// fp32 underflow: exp(log_iv) -> 0, log(1e-10)), assemble the 4 outputs.
// ---------------------------------------------------------------------------
__global__ void vmf_finalize_kernel(const float* __restrict__ kappa_p,
                                    const float* __restrict__ acc,
                                    float* __restrict__ out) {
    if (threadIdx.x != 0 || blockIdx.x != 0) return;
    const float nu  = DIM * 0.5f - 1.0f;
    const float kap = fmaxf(kappa_p[0], 0.001f);
    const float ks  = fmaxf(kap, 1e-10f);
    const float lx  = logf(ks * 0.5f);

    float tmax = -INFINITY;
    for (int k = 0; k < 64; ++k) {
        float fk = (float)k;
        float tv = (nu + 2.0f * fk) * lx - lgammaf(fk + 1.0f) - lgammaf(nu + fk + 1.0f);
        tmax = fmaxf(tmax, tv);
    }
    float se = 0.0f;
    for (int k = 0; k < 64; ++k) {
        float fk = (float)k;
        float tv = (nu + 2.0f * fk) * lx - lgammaf(fk + 1.0f) - lgammaf(nu + fk + 1.0f);
        se += expf(tv - tmax);
    }
    float log_iv     = tmax + logf(se);
    float iv         = expf(log_iv);            // underflows like the fp32 reference
    float log_bessel = logf(iv + 1e-10f);
    float log_C = nu * logf(kap) - (DIM * 0.5f) * logf(2.0f * (float)M_PI) - log_bessel;

    float inv_n    = 1.0f / (float)NROWS;
    float mean_dot = acc[2] * inv_n;
    float vmf      = -(log_C + kap * mean_dot);
    float mean_n   = acc[0] * inv_n;
    float radius   = acc[1] * inv_n - mean_n * mean_n;
    float rep      = acc[3] / ((float)NROWS * (float)(NROWS - 1));

    float rw = 0.8f * radius;
    float pw = 0.2f * rep;
    out[0] = vmf + rw + pw;
    out[1] = vmf;
    out[2] = rw;
    out[3] = pw;
}

// ---------------------------------------------------------------------------
extern "C" void kernel_launch(void* const* d_in, const int* in_sizes, int n_in,
                              void* d_out, int out_size, void* d_ws, size_t ws_size,
                              hipStream_t stream) {
    const float* x     = (const float*)d_in[0];
    const float* mu    = (const float*)d_in[1];
    const float* kappa = (const float*)d_in[2];
    float* out = (float*)d_out;

    char* ws = (char*)d_ws;
    __hip_bfloat16* xu = (__hip_bfloat16*)ws;                       // 8 MB bf16 unit rows
    float* nsq  = (float*)(ws + (size_t)NROWS * DIM * 2);           // 32 KB
    float* mu_n = (float*)(ws + (size_t)NROWS * DIM * 2 + NROWS * 4);
    float* acc  = (float*)(ws + (size_t)NROWS * DIM * 2 + NROWS * 4 + DIM * 4);

    vmf_init_kernel<<<1, 512, 0, stream>>>(mu, mu_n, acc);
    vmf_prep_kernel<<<NROWS, 256, 0, stream>>>(x, mu_n, xu, nsq, acc);

    const int n_waves = NST * NST;                                  // 16384 (half exit early)
    vmf_gram_kernel<<<n_waves / WAVES_PER_BLOCK, WAVES_PER_BLOCK * 32, 0, stream>>>(xu, nsq, acc);

    vmf_finalize_kernel<<<1, 1, 0, stream>>>(kappa, acc, out);
}
